// MultiGATBase_65214783422917
// MI455X (gfx1250) — matrix operand order repaired and checked by reading, not compiled, not run on hardware
//
#include <hip/hip_runtime.h>
#include <cstdint>
#include <cstddef>

// ---------------- problem constants (from reference) ----------------
#define N_NODES   2048
#define N_EDGES   32768
#define HEAD_D    512
#define NEG_SLOPE 0.01f

// ---------------- vector types ----------------
typedef __attribute__((ext_vector_type(16))) __bf16 v16bf;
typedef __attribute__((ext_vector_type(8)))  __bf16 v8bf;
typedef __attribute__((ext_vector_type(4)))  __bf16 v4bf;
typedef __attribute__((ext_vector_type(8)))  float  v8f;

// pointer-to-int4 types matching the async-LDS builtin's signature
typedef int v4i_vs __attribute__((vector_size(16)));
typedef __attribute__((address_space(1))) v4i_vs* gptr128_t;
typedef __attribute__((address_space(3))) v4i_vs* lptr128_t;

static __device__ __forceinline__ __bf16 f2bf(float f) {
  // round-to-nearest-even fp32 -> bf16
  unsigned u = __float_as_uint(f);
  unsigned r = (u + 0x7FFFu + ((u >> 16) & 1u)) >> 16;
  unsigned short s = (unsigned short)r;
  return __builtin_bit_cast(__bf16, s);
}

// ---------------- gfx1250 async global->LDS copy (ASYNCcnt path) ----------------
#if __has_builtin(__builtin_amdgcn_global_load_async_to_lds_b128)
#define HAS_ASYNC_LDS 1
#else
#define HAS_ASYNC_LDS 0
#endif

static __device__ __forceinline__ void copy16_g2l(const __bf16* g, __bf16* l) {
#if HAS_ASYNC_LDS
  __builtin_amdgcn_global_load_async_to_lds_b128((gptr128_t)g, (lptr128_t)l, 0, 0);
#else
  *(v8bf*)l = *(const v8bf*)g;
#endif
}

static __device__ __forceinline__ void async_wait_all() {
#if HAS_ASYNC_LDS
#if __has_builtin(__builtin_amdgcn_s_wait_asynccnt)
  __builtin_amdgcn_s_wait_asynccnt(0);
#else
  asm volatile("s_wait_asynccnt 0x0" ::: "memory");
#endif
#endif
}

// =====================================================================
// Weight transpose + convert:  W fp32 [H][K][512]  ->  Wt bf16 [H][512][K]
// (so GEMM "B" is row-major over output columns: row n holds K contiguous
// bf16 values -> staging is a raw 16B copy, fragments are ds_load_b128)
// =====================================================================
__global__ __launch_bounds__(256)
void gat_w_transpose_bf16(const float* __restrict__ W, __bf16* __restrict__ Wt, int K) {
  __shared__ float t[32][33];
  const int h  = blockIdx.z;
  const int k0 = blockIdx.x * 32;
  const int d0 = blockIdx.y * 32;
  const float* Wb = W  + ((size_t)h * K + k0) * HEAD_D + d0;
  __bf16*      Wo = Wt + ((size_t)h * HEAD_D + d0) * K + k0;
  const int tx = threadIdx.x & 31;
  const int ty = threadIdx.x >> 5;   // 0..7
  #pragma unroll
  for (int p = 0; p < 4; ++p)
    t[ty + 8 * p][tx] = Wb[(size_t)(ty + 8 * p) * HEAD_D + tx];
  __syncthreads();
  #pragma unroll
  for (int p = 0; p < 4; ++p)
    Wo[(size_t)(ty + 8 * p) * K + tx] = f2bf(t[tx][ty + 8 * p]);
}

// elementwise fp32 -> bf16 (for the input features)
__global__ void gat_f32_to_bf16(const float* __restrict__ in, __bf16* __restrict__ out, int n) {
  int i = blockIdx.x * blockDim.x + threadIdx.x;
  if (i < n) out[i] = f2bf(in[i]);
}

// =====================================================================
// GEMM:  C[M, HD] = A[M, K] * Bt^T,  A bf16 row-major [M][K],
// Bt bf16 row-major [HD][K] (pre-transposed weights).
// bf16 WMMA 16x16x32, f32 accumulate. Block tile 128x128, 256 thr = 8 waves
// (2M x 4N), wave tile 64x32 = 4x2 accumulators. Double-buffered LDS with
// async global->LDS copies overlapping the WMMAs.
// =====================================================================
#define BM 128
#define BN 128
#define BK 32
#define LDSS 40   // bf16 elements per LDS row (80 B, multiple of 16 B)

__global__ __launch_bounds__(256)
void gat_gemm_bf16_wmma(const __bf16* __restrict__ A,
                        const __bf16* __restrict__ Bt,
                        float* __restrict__ C,
                        int K, int HD) {
  __shared__ __attribute__((aligned(16))) __bf16 As[2][BM * LDSS];
  __shared__ __attribute__((aligned(16))) __bf16 Bs[2][BN * LDSS];

  const int m0 = blockIdx.x * BM;   // 16 M-tiles: consecutive blocks share B panel
  const int n0 = blockIdx.y * BN;

  const int tid    = threadIdx.x;
  const int wave   = tid >> 5;
  const int lane   = tid & 31;
  const int laneLo = lane & 15;
  const int hi     = lane >> 4;        // half-wave select
  const int wrow   = (wave & 1) * 64;  // wave M offset
  const int wcol   = (wave >> 1) * 32; // wave N offset

  // ---- staging assignment: 512 chunks of 16B per tile, 2 per thread ----
  const int srow = tid >> 2;           // 0..63
  const int soff = (tid & 3) * 8;      // bf16 element offset within row
  const __bf16* gA0 = A  + (size_t)(m0 + srow) * K + soff;
  const __bf16* gA1 = gA0 + (size_t)64 * K;
  const __bf16* gB0 = Bt + (size_t)(n0 + srow) * K + soff;
  const __bf16* gB1 = gB0 + (size_t)64 * K;
  const int l0 = srow * LDSS + soff;
  const int l1 = l0 + 64 * LDSS;

  v8f acc[4][2];
  v8f vzero;
  #pragma unroll
  for (int e = 0; e < 8; ++e) vzero[e] = 0.0f;
  #pragma unroll
  for (int i = 0; i < 4; ++i)
    #pragma unroll
    for (int j = 0; j < 2; ++j) acc[i][j] = vzero;

  // prime buffer 0
  copy16_g2l(gA0, &As[0][l0]);
  copy16_g2l(gA1, &As[0][l1]);
  copy16_g2l(gB0, &Bs[0][l0]);
  copy16_g2l(gB1, &Bs[0][l1]);
  async_wait_all();
  __syncthreads();

  int cur = 0;
  for (int k0 = 0; k0 < K; k0 += BK) {
    const int nxt = cur ^ 1;
    const bool more = (k0 + BK) < K;
    if (more) {
      const int kn = k0 + BK;
      copy16_g2l(gA0 + kn, &As[nxt][l0]);
      copy16_g2l(gA1 + kn, &As[nxt][l1]);
      copy16_g2l(gB0 + kn, &Bs[nxt][l0]);
      copy16_g2l(gB1 + kn, &Bs[nxt][l1]);
    }

    // ---- fragments from current buffer ----
    // A 16x32 bf16: lanes 0-15 -> K0..7 & K16..23 ; lanes 16-31 -> K8..15 & K24..31
    v16bf afrag[4], bfrag[2];
    const __bf16* Ab = As[cur];
    const __bf16* Bb = Bs[cur];
    #pragma unroll
    for (int i = 0; i < 4; ++i) {
      const __bf16* rp = Ab + (wrow + i * 16 + laneLo) * LDSS;
      v8bf lo = *(const v8bf*)(rp + hi * 8);
      v8bf hv = *(const v8bf*)(rp + 16 + hi * 8);
      afrag[i] = __builtin_shufflevector(lo, hv, 0,1,2,3,4,5,6,7,8,9,10,11,12,13,14,15);
    }
    // B 32x16 bf16: lane = column; lanes 0-15 -> K0..7 & K16..23 ; lanes 16-31 -> K8..15 & K24..31
    #pragma unroll
    for (int j = 0; j < 2; ++j) {
      const __bf16* rp = Bb + (wcol + j * 16 + laneLo) * LDSS;
      v8bf lo = *(const v8bf*)(rp + hi * 8);
      v8bf hv = *(const v8bf*)(rp + 16 + hi * 8);
      bfrag[j] = __builtin_shufflevector(lo, hv, 0,1,2,3,4,5,6,7,8,9,10,11,12,13,14,15);
    }

    #pragma unroll
    for (int i = 0; i < 4; ++i)
      #pragma unroll
      for (int j = 0; j < 2; ++j)
        acc[i][j] = __builtin_amdgcn_wmma_f32_16x16x32_bf16(
            false, afrag[i], false, bfrag[j], (short)0, acc[i][j], false, false);

    if (more) async_wait_all();
    __syncthreads();
    cur = nxt;
  }

  // ---- store C (f32): lanes 0-15 rows v, lanes 16-31 rows v+8 ----
  #pragma unroll
  for (int i = 0; i < 4; ++i)
    #pragma unroll
    for (int j = 0; j < 2; ++j)
      #pragma unroll
      for (int v = 0; v < 8; ++v) {
        int row = m0 + wrow + i * 16 + hi * 8 + v;
        int col = n0 + wcol + j * 16 + laneLo;
        C[(size_t)row * HD + col] = acc[i][j][v];
      }
}

// =====================================================================
// s_src[n,h] = z[n,h,:] . a[h,0:512];  s_dst[n,h] = z[n,h,:] . a[h,512:1024]
// =====================================================================
__global__ __launch_bounds__(256)
void gat_scores(const float* __restrict__ z, const float* __restrict__ a,
                float* __restrict__ s_src, float* __restrict__ s_dst, int H) {
  const int b = blockIdx.x;
  const int h = b % H;
  const float* zr   = z + (size_t)b * HEAD_D;
  const float* asrc = a + (size_t)h * (2 * HEAD_D);
  const float* adst = asrc + HEAD_D;
  const int t = threadIdx.x;

  float accS = zr[t] * asrc[t] + zr[t + 256] * asrc[t + 256];
  float accD = zr[t] * adst[t] + zr[t + 256] * adst[t + 256];

  #pragma unroll
  for (int off = 16; off > 0; off >>= 1) {
    accS += __shfl_down(accS, off, 32);
    accD += __shfl_down(accD, off, 32);
  }
  __shared__ float rs[8], rd[8];
  const int wv = t >> 5;
  if ((t & 31) == 0) { rs[wv] = accS; rd[wv] = accD; }
  __syncthreads();
  if (t == 0) {
    float s = 0.f, d = 0.f;
    #pragma unroll
    for (int w = 0; w < 8; ++w) { s += rs[w]; d += rd[w]; }
    s_src[b] = s;
    s_dst[b] = d;
  }
}

// =====================================================================
// CSR build over dst (graph fixed for all 4 layers)
// =====================================================================
__global__ void gat_zero_i32(int* p, int n) {
  int i = blockIdx.x * blockDim.x + threadIdx.x;
  if (i < n) p[i] = 0;
}
__global__ void gat_csr_count(const int* __restrict__ dst, int* __restrict__ counts) {
  int e = blockIdx.x * blockDim.x + threadIdx.x;
  if (e < N_EDGES) atomicAdd(&counts[dst[e] + 1], 1);
}
__global__ void gat_csr_scan(int* __restrict__ counts, int* __restrict__ cursor) {
  if (blockIdx.x == 0 && threadIdx.x == 0) {
    int run = 0;
    for (int i = 0; i <= N_NODES; ++i) { run += counts[i]; counts[i] = run; }
    for (int i = 0; i < N_NODES; ++i) cursor[i] = counts[i];
  }
}
__global__ void gat_csr_fill(const int* __restrict__ dst, int* __restrict__ cursor,
                             int* __restrict__ eids) {
  int e = blockIdx.x * blockDim.x + threadIdx.x;
  if (e < N_EDGES) {
    int pos = atomicAdd(&cursor[dst[e]], 1);
    eids[pos] = e;
  }
}

// =====================================================================
// Per (dst node, head) softmax over incoming edges + gather-aggregate.
// deg ~ 16: every thread redundantly walks the edge list for max/sum-exp
// (scalar-cached loads), then 128 threads x float4 accumulate D=512 with
// no atomics. Output: bf16 (feeds next layer's GEMM A) or f32 (final).
// =====================================================================
static __device__ __forceinline__ float lrelu(float x) {
  return x > 0.f ? x : x * NEG_SLOPE;
}

template <bool BF16OUT>
__global__ __launch_bounds__(128)
void gat_aggregate(const float* __restrict__ z,
                   const float* __restrict__ s_src,
                   const float* __restrict__ s_dst,
                   const int* __restrict__ offs,
                   const int* __restrict__ eids,
                   const int* __restrict__ srcIdx,
                   float* __restrict__ outF,
                   __bf16* __restrict__ outB, int H) {
  const int n = blockIdx.x;
  const int h = blockIdx.y;
  const int HD = H * HEAD_D;
  const int start = offs[n], end = offs[n + 1];
  const int t = threadIdx.x;
  const size_t obase = (size_t)n * HD + (size_t)h * HEAD_D;

  float4 acc = make_float4(0.f, 0.f, 0.f, 0.f);
  if (end > start) {
    const float sd = s_dst[n * H + h];
    float m = -3.402823466e38f;
    for (int i = start; i < end; ++i) {
      int e = eids[i];
      m = fmaxf(m, lrelu(s_src[srcIdx[e] * H + h] + sd));
    }
    float dn = 0.f;
    for (int i = start; i < end; ++i) {
      int e = eids[i];
      dn += __expf(lrelu(s_src[srcIdx[e] * H + h] + sd) - m);
    }
    const float inv = 1.0f / dn;
    for (int i = start; i < end; ++i) {
      int e = eids[i];
      int s = srcIdx[e];
      float alpha = __expf(lrelu(s_src[s * H + h] + sd) - m) * inv;
      const float4 v = ((const float4*)(z + (size_t)s * HD + (size_t)h * HEAD_D))[t];
      acc.x += alpha * v.x; acc.y += alpha * v.y;
      acc.z += alpha * v.z; acc.w += alpha * v.w;
    }
  }
  if (BF16OUT) {
    v4bf o;
    o[0] = f2bf(acc.x); o[1] = f2bf(acc.y); o[2] = f2bf(acc.z); o[3] = f2bf(acc.w);
    *(v4bf*)(outB + obase + (size_t)t * 4) = o;
  } else {
    ((float4*)(outF + obase))[t] = acc;
  }
}

// =====================================================================
// host side
// =====================================================================
static void run_gat_layer(const __bf16* xbf, int Fin, int H,
                          const float* W, const float* a,
                          __bf16* wbf, float* zbuf,
                          float* outF, __bf16* outB,
                          float* ssrc, float* sdst,
                          const int* offs, const int* eids, const int* srcIdx,
                          hipStream_t stream) {
  const int HD = H * HEAD_D;
  // transpose+convert weights: [H][Fin][512] -> bf16 [H][512][Fin]
  gat_w_transpose_bf16<<<dim3(Fin / 32, HEAD_D / 32, H), 256, 0, stream>>>(W, wbf, Fin);
  gat_gemm_bf16_wmma<<<dim3(N_NODES / BM, HD / BN), 256, 0, stream>>>(xbf, wbf, zbuf, Fin, HD);
  gat_scores<<<N_NODES * H, 256, 0, stream>>>(zbuf, a, ssrc, sdst, H);
  if (outB)
    gat_aggregate<true><<<dim3(N_NODES, H), 128, 0, stream>>>(
        zbuf, ssrc, sdst, offs, eids, srcIdx, nullptr, outB, H);
  else
    gat_aggregate<false><<<dim3(N_NODES, H), 128, 0, stream>>>(
        zbuf, ssrc, sdst, offs, eids, srcIdx, outF, nullptr, H);
}

extern "C" void kernel_launch(void* const* d_in, const int* in_sizes, int n_in,
                              void* d_out, int out_size, void* d_ws, size_t ws_size,
                              hipStream_t stream) {
  const float* feat = (const float*)d_in[0];
  const float* W1 = (const float*)d_in[1]; const float* a1 = (const float*)d_in[2];
  const float* W2 = (const float*)d_in[3]; const float* a2 = (const float*)d_in[4];
  const float* W3 = (const float*)d_in[5]; const float* a3 = (const float*)d_in[6];
  const float* W4 = (const float*)d_in[7]; const float* a4 = (const float*)d_in[8];
  const int* src  = (const int*)d_in[9];
  const int* dst  = (const int*)d_in[10];

  // ---- workspace layout ----
  const size_t bigN = (size_t)N_NODES * 16 * HEAD_D;        // 2048*8192
  const size_t wMax = (size_t)16 * HEAD_D * 8192;           // largest Wt (bf16 elems)
  float*  bufZ = (float*)d_ws;                              // 64 MB fp32 z
  __bf16* hbf  = (__bf16*)(bufZ + bigN);                    // 32 MB bf16 activations
  __bf16* wbf  = hbf + bigN;                                // 128 MB bf16 weights
  float*  ssrc = (float*)(wbf + wMax);
  float*  sdst = ssrc + (size_t)N_NODES * 16;
  int*    counts = (int*)(sdst + (size_t)N_NODES * 16);     // N_NODES+1 offsets
  int*    cursor = counts + (N_NODES + 16);
  int*    eids   = cursor + N_NODES;

  // ---- CSR build (graph fixed across layers) ----
  gat_zero_i32<<<(N_NODES + 1 + 255) / 256, 256, 0, stream>>>(counts, N_NODES + 1);
  gat_csr_count<<<N_EDGES / 256, 256, 0, stream>>>(dst, counts);
  gat_csr_scan<<<1, 1, 0, stream>>>(counts, cursor);
  gat_csr_fill<<<N_EDGES / 256, 256, 0, stream>>>(dst, cursor, eids);

  // ---- input features -> bf16 ----
  gat_f32_to_bf16<<<(N_NODES * 512 + 255) / 256, 256, 0, stream>>>(feat, hbf, N_NODES * 512);

  // ---- 4 GAT layers (hbf ping-pongs as bf16 activation buffer) ----
  run_gat_layer(hbf, 512,  16, W1, a1, wbf, bufZ, nullptr, hbf, ssrc, sdst, counts, eids, src, stream);
  run_gat_layer(hbf, 8192, 16, W2, a2, wbf, bufZ, nullptr, hbf, ssrc, sdst, counts, eids, src, stream);
  run_gat_layer(hbf, 8192, 16, W3, a3, wbf, bufZ, nullptr, hbf, ssrc, sdst, counts, eids, src, stream);
  run_gat_layer(hbf, 8192, 1,  W4, a4, wbf, bufZ, (float*)d_out, nullptr, ssrc, sdst, counts, eids, src, stream);
}
